// EVADRoIHead_60215441489978
// MI455X (gfx1250) — compile-verified, matrix operand order checked by reading, not run on hardware
//
#include <hip/hip_runtime.h>
#include <hip/hip_bf16.h>

// ---------------------------------------------------------------------------
// Types / helpers
// ---------------------------------------------------------------------------
typedef __attribute__((ext_vector_type(16))) __bf16 v16bf;
typedef __attribute__((ext_vector_type(8)))  __bf16 v8bf;
typedef __attribute__((ext_vector_type(8)))  float  v8f;

__device__ __forceinline__ unsigned short f2bfu(float f) {
    return __builtin_bit_cast(unsigned short, (__bf16)f);   // HW RNE convert
}

// A/B fragment (16x32 bf16, wave32). Per ISA 7.12.2:
// lane = half*16 + r ; row = (m|n)+r ; elems 0..7  <- K = kb+half*8+[0..7],
// elems 8..15 <- K = kb+16+half*8+[0..7]   (K-contiguous per lane).
__device__ __forceinline__ v16bf load_frag(const float* rowp, int kb, int half) {
    const float4* p = reinterpret_cast<const float4*>(rowp + kb + half * 8);
    float4 x0 = p[0], x1 = p[1];   // K + 0..7
    float4 y0 = p[4], y1 = p[5];   // K + 16..23
    v16bf f;
    f[0] = (__bf16)x0.x; f[1] = (__bf16)x0.y; f[2]  = (__bf16)x0.z; f[3]  = (__bf16)x0.w;
    f[4] = (__bf16)x1.x; f[5] = (__bf16)x1.y; f[6]  = (__bf16)x1.z; f[7]  = (__bf16)x1.w;
    f[8] = (__bf16)y0.x; f[9] = (__bf16)y0.y; f[10] = (__bf16)y0.z; f[11] = (__bf16)y0.w;
    f[12]= (__bf16)y1.x; f[13]= (__bf16)y1.y; f[14] = (__bf16)y1.z; f[15] = (__bf16)y1.w;
    return f;
}
__device__ __forceinline__ v16bf load_frag_masked(const float* rowp, int kb, int half, float m) {
    const float4* p = reinterpret_cast<const float4*>(rowp + kb + half * 8);
    float4 x0 = p[0], x1 = p[1];
    float4 y0 = p[4], y1 = p[5];
    v16bf f;
    f[0] = (__bf16)(x0.x*m); f[1] = (__bf16)(x0.y*m); f[2]  = (__bf16)(x0.z*m); f[3]  = (__bf16)(x0.w*m);
    f[4] = (__bf16)(x1.x*m); f[5] = (__bf16)(x1.y*m); f[6]  = (__bf16)(x1.z*m); f[7]  = (__bf16)(x1.w*m);
    f[8] = (__bf16)(y0.x*m); f[9] = (__bf16)(y0.y*m); f[10] = (__bf16)(y0.z*m); f[11] = (__bf16)(y0.w*m);
    f[12]= (__bf16)(y1.x*m); f[13]= (__bf16)(y1.y*m); f[14] = (__bf16)(y1.z*m); f[15] = (__bf16)(y1.w*m);
    return f;
}
// bf16 source (global or LDS): two 16-byte vector loads, zero ALU.
__device__ __forceinline__ v16bf load_frag(const unsigned short* rowp, int kb, int half) {
    const v8bf* p = reinterpret_cast<const v8bf*>(rowp + kb + half * 8);
    v8bf lo = p[0];       // K + 0..7
    v8bf hi = p[2];       // K + 16..23
    return __builtin_shufflevector(lo, hi, 0,1,2,3,4,5,6,7,8,9,10,11,12,13,14,15);
}

__device__ __forceinline__ void storeC(float* C, size_t idx, float v) { C[idx] = v; }
__device__ __forceinline__ void storeC(unsigned short* C, size_t idx, float v) { C[idx] = f2bfu(v); }

#define WMMA_BF16(a, b, c) \
    __builtin_amdgcn_wmma_f32_16x16x32_bf16(false, (a), false, (b), (short)0, (c), false, false)

// ---------------------------------------------------------------------------
// Generic WMMA GEMM:  C(MxN) = A(MxK) @ Bw(NxK)^T + bias,  optional ReLU.
// block = 128 threads (4 waves); block tile 64(M) x 64(N); wave tile 64x16.
// M % 64 == 0, N % 64 == 0, K % 32 == 0 (holds for every call site here).
// ---------------------------------------------------------------------------
template <typename AT, typename CT, bool RELU>
__global__ __launch_bounds__(128) void wmma_gemm(const AT* __restrict__ A,
                                                 const float* __restrict__ Bw,
                                                 const float* __restrict__ bias,
                                                 CT* __restrict__ C,
                                                 int M, int N, int K) {
    const int lane = threadIdx.x & 31;
    const int wv   = threadIdx.x >> 5;
    const int half = lane >> 4;
    const int r    = lane & 15;
    const int n    = blockIdx.y * 64 + wv * 16 + r;
    const int mb   = blockIdx.x * 64;

    const float* brow = Bw + (size_t)n * K;
    const AT* a0 = A + (size_t)(mb +  0 + r) * K;
    const AT* a1 = A + (size_t)(mb + 16 + r) * K;
    const AT* a2 = A + (size_t)(mb + 32 + r) * K;
    const AT* a3 = A + (size_t)(mb + 48 + r) * K;

    v8f acc[4] = {{}, {}, {}, {}};
    for (int kb = 0; kb < K; kb += 32) {
        v16bf bf = load_frag(brow, kb, half);
        acc[0] = WMMA_BF16(load_frag(a0, kb, half), bf, acc[0]);
        acc[1] = WMMA_BF16(load_frag(a1, kb, half), bf, acc[1]);
        acc[2] = WMMA_BF16(load_frag(a2, kb, half), bf, acc[2]);
        acc[3] = WMMA_BF16(load_frag(a3, kb, half), bf, acc[3]);
    }
    const float bv = bias ? bias[n] : 0.0f;
#pragma unroll
    for (int s = 0; s < 4; ++s) {
#pragma unroll
        for (int vr = 0; vr < 8; ++vr) {
            int m = mb + s * 16 + vr + half * 8;       // C layout: VGPR vr -> M=vr / vr+8
            float v = acc[s][vr] + bv;
            if (RELU) v = fmaxf(v, 0.0f);
            storeC(C, (size_t)m * N + n, v);
        }
    }
}

// ---------------------------------------------------------------------------
// Per-instance dynamic conv (one block / instance, 8 waves):
//  p1 (256x64) staged transposed into LDS; t1 = roi_b(49x256) @ p1 (WMMA)
//  -> LN(dn1)+ReLU -> bf16 LDS; p2 (64x256) staged transposed into LDS;
//  t2 = t1(49x64) @ p2 (WMMA) -> LN(dn2)+ReLU -> bf16 global.
// LDS: [p1T|p2T 32KB][sA 16KB / sC 64KB][t1b 8KB] = 106496 B (aliased).
// sA/sC padded to 64 rows so WMMA epilogues store unconditionally
// (rows 49..63 are scratch, never read by the LayerNorm passes).
// ---------------------------------------------------------------------------
__global__ __launch_bounds__(256) void dynconv_kernel(const float* __restrict__ roi,
                                                      const unsigned short* __restrict__ dynC,
                                                      const float* __restrict__ g1, const float* __restrict__ b1,
                                                      const float* __restrict__ g2, const float* __restrict__ b2,
                                                      unsigned short* __restrict__ f2out,
                                                      int inst_base) {
    __shared__ __align__(16) unsigned char smem[106496];
    unsigned short* p1T = (unsigned short*)smem;          // stage 1: p1^T (64 e x 256 k)
    unsigned short* p2T = (unsigned short*)smem;          // stage 2: p2^T (256 d x 64 e)
    float* sA  = (float*)(smem + 32768);                  // stage-1 out 64x64 (49 valid)
    float* sC  = (float*)(smem + 32768);                  // stage-2 out 64x256 (49 valid)
    unsigned short* t1b = (unsigned short*)(smem + 98304);// LN1 out, bf16 64x64

    const int b_loc = blockIdx.x;
    const int b     = inst_base + b_loc;
    const float* roiB = roi + (size_t)b * 49 * 256;
    const unsigned short* p1 = dynC + (size_t)b_loc * 32768;
    const unsigned short* p2 = p1 + 16384;

    const int tid  = threadIdx.x;
    const int lane = tid & 31;
    const int wv   = tid >> 5;
    const int half = lane >> 4;
    const int r    = lane & 15;

    // ---- stage p1 -> LDS transposed: p1T[e*256+k] -----------------------
    {
        const unsigned short* src = p1 + tid * 64;        // k = tid, e = 0..63
        unsigned short tmp[64];
#pragma unroll
        for (int i = 0; i < 8; ++i)
            ((uint4*)tmp)[i] = ((const uint4*)src)[i];
#pragma unroll
        for (int e = 0; e < 64; ++e) p1T[e * 256 + tid] = tmp[e];
    }
    __syncthreads();

    // ---- stage 1: 49x64 = roi @ p1 --------------------------------------
    for (int t = wv; t < 16; t += 8) {
        const int mt = t >> 2, nt = t & 3;
        const int mrow = mt * 16 + r;
        const int n = nt * 16 + r;
        const float msk = (mrow < 49) ? 1.0f : 0.0f;
        const float* ap = roiB + (size_t)(mrow < 49 ? mrow : 0) * 256;
        const unsigned short* bp = p1T + n * 256;
        v8f acc = {};
        for (int kb = 0; kb < 256; kb += 32)
            acc = WMMA_BF16(load_frag_masked(ap, kb, half, msk),
                            load_frag(bp, kb, half), acc);
#pragma unroll
        for (int vr = 0; vr < 8; ++vr) {
            int m = mt * 16 + vr + half * 8;
            sA[m * 64 + n] = acc[vr];                     // padded, unconditional
        }
    }
    __syncthreads();

    // ---- LN(dn1)+ReLU -> t1b ; stage p2 -> LDS transposed ---------------
    if (tid < 49) {
        float s = 0.f;
        for (int e = 0; e < 64; ++e) s += sA[tid * 64 + e];
        float mean = s * (1.0f / 64.0f), v = 0.f;
        for (int e = 0; e < 64; ++e) { float d = sA[tid * 64 + e] - mean; v += d * d; }
        float inv = rsqrtf(v * (1.0f / 64.0f) + 1e-5f);
        for (int e = 0; e < 64; ++e) {
            float y = (sA[tid * 64 + e] - mean) * inv * g1[e] + b1[e];
            t1b[tid * 64 + e] = f2bfu(fmaxf(y, 0.0f));
        }
    } else if (tid < 64) {
        for (int e = 0; e < 64; ++e) t1b[tid * 64 + e] = 0;
    }
    {   // p2 (64 e x 256 d) -> p2T[d*64+e]; thread: e = tid/4, d block of 64
        const int e  = tid >> 2;
        const int d0 = (tid & 3) * 64;
        const unsigned short* src = p2 + e * 256 + d0;
        unsigned short tmp[64];
#pragma unroll
        for (int i = 0; i < 8; ++i)
            ((uint4*)tmp)[i] = ((const uint4*)src)[i];
        __syncthreads();                                  // sA consumed, p1T dead
#pragma unroll
        for (int j = 0; j < 64; ++j) p2T[(d0 + j) * 64 + e] = tmp[j];
    }
    __syncthreads();

    // ---- stage 2: 49x256 = t1 @ p2 --------------------------------------
    for (int t = wv; t < 64; t += 8) {
        const int mt = t >> 4, nt = t & 15;
        const int n = nt * 16 + r;
        const unsigned short* ap = t1b + (mt * 16 + r) * 64;
        const unsigned short* bp = p2T + n * 64;
        v8f acc = {};
        for (int kb = 0; kb < 64; kb += 32)
            acc = WMMA_BF16(load_frag(ap, kb, half),
                            load_frag(bp, kb, half), acc);
#pragma unroll
        for (int vr = 0; vr < 8; ++vr) {
            int m = mt * 16 + vr + half * 8;
            sC[m * 256 + n] = acc[vr];                    // padded, unconditional
        }
    }
    __syncthreads();

    // ---- LN(dn2)+ReLU over 256, bf16 to global --------------------------
    if (tid < 49) {
        float s = 0.f;
        for (int d = 0; d < 256; ++d) s += sC[tid * 256 + d];
        float mean = s * (1.0f / 256.0f), v = 0.f;
        for (int d = 0; d < 256; ++d) { float e = sC[tid * 256 + d] - mean; v += e * e; }
        float inv = rsqrtf(v * (1.0f / 256.0f) + 1e-5f);
        size_t off = ((size_t)b * 49 + tid) * 256;
        for (int d = 0; d < 256; ++d) {
            float y = (sC[tid * 256 + d] - mean) * inv * g2[d] + b2[d];
            f2out[off + d] = f2bfu(fmaxf(y, 0.0f));
        }
    }
}

// ---------------------------------------------------------------------------
// Small VALU kernels
// ---------------------------------------------------------------------------
__global__ void permute_kernel(const float* __restrict__ pro, float* __restrict__ x) {
    int row = blockIdx.x;                 // dst row = s*16 + b
    int s = row >> 4, b = row & 15;
    x[(size_t)row * 256 + threadIdx.x] = pro[((size_t)b * 100 + s) * 256 + threadIdx.x];
}

__global__ __launch_bounds__(128) void attn_kernel(const float* __restrict__ qkv,
                                                   float* __restrict__ o) {
    const int s = blockIdx.x, h = blockIdx.y, b = blockIdx.z;
    const int tid = threadIdx.x;
    __shared__ float sc[100];
    __shared__ float qs[32];
    __shared__ float red[2];
    if (tid < 32) qs[tid] = qkv[((size_t)(s * 16 + b)) * 768 + h * 32 + tid];
    __syncthreads();
    if (tid < 100) {
        const float* kk = qkv + ((size_t)(tid * 16 + b)) * 768 + 256 + h * 32;
        float d = 0.f;
#pragma unroll
        for (int j = 0; j < 32; ++j) d += qs[j] * kk[j];
        sc[tid] = d * 0.17677669529663687f;   // 1/sqrt(32)
    }
    __syncthreads();
    if (tid == 0) {
        float m = -1e30f;
        for (int t = 0; t < 100; ++t) m = fmaxf(m, sc[t]);
        red[0] = m;
    }
    __syncthreads();
    if (tid < 100) sc[tid] = __expf(sc[tid] - red[0]);
    __syncthreads();
    if (tid == 0) {
        float su = 0.f;
        for (int t = 0; t < 100; ++t) su += sc[t];
        red[1] = 1.0f / su;
    }
    __syncthreads();
    if (tid < 32) {
        float acc = 0.f;
        for (int t = 0; t < 100; ++t)
            acc += sc[t] * qkv[((size_t)(t * 16 + b)) * 768 + 512 + h * 32 + tid];
        o[((size_t)(s * 16 + b)) * 256 + h * 32 + tid] = acc * red[1];
    }
}

template <bool RES, bool RELU, bool PERM>
__global__ __launch_bounds__(256) void ln_kernel(const float* __restrict__ X,
                                                 const float* __restrict__ R,
                                                 const float* __restrict__ g,
                                                 const float* __restrict__ bta,
                                                 float* __restrict__ out) {
    __shared__ float red[256];
    const int dst = blockIdx.x, tid = threadIdx.x;
    const int src = PERM ? ((dst % 100) * 16 + dst / 100) : dst;
    float val = X[(size_t)src * 256 + tid];
    if (RES) val += R[(size_t)src * 256 + tid];
    red[tid] = val;
    __syncthreads();
    for (int off = 128; off > 0; off >>= 1) { if (tid < off) red[tid] += red[tid + off]; __syncthreads(); }
    float mean = red[0] * (1.0f / 256.0f);
    __syncthreads();
    float d = val - mean;
    red[tid] = d * d;
    __syncthreads();
    for (int off = 128; off > 0; off >>= 1) { if (tid < off) red[tid] += red[tid + off]; __syncthreads(); }
    float var = red[0] * (1.0f / 256.0f);
    float y = d * rsqrtf(var + 1e-5f) * g[tid] + bta[tid];
    if (RELU) y = fmaxf(y, 0.0f);
    out[(size_t)dst * 256 + tid] = y;
}

__global__ __launch_bounds__(256) void head_kernel(const float* __restrict__ X,
                                                   const float* __restrict__ W,
                                                   const float* __restrict__ bias,
                                                   float* __restrict__ out, int ncols) {
    __shared__ float sx[256];
    const int row = blockIdx.x, tid = threadIdx.x;
    sx[tid] = X[(size_t)row * 256 + tid];
    __syncthreads();
    if (tid < ncols) {
        float s = bias[tid];
        for (int k = 0; k < 256; ++k) s += sx[k] * W[tid * 256 + k];
        out[(size_t)row * ncols + tid] = s;
    }
}

__global__ void apply_deltas_kernel(const float* __restrict__ boxes,
                                    const float* __restrict__ del,
                                    float* __restrict__ pred) {
    int i = blockIdx.x * blockDim.x + threadIdx.x;
    if (i >= 1600) return;
    const float SCALE_CLAMP = 8.740336742730447f;   // log(100000/16)
    float x0 = boxes[i * 4 + 0], y0 = boxes[i * 4 + 1];
    float x1 = boxes[i * 4 + 2], y1 = boxes[i * 4 + 3];
    float w = x1 - x0, h = y1 - y0;
    float cx = x0 + 0.5f * w, cy = y0 + 0.5f * h;
    float dx = del[i * 4 + 0] * 0.5f, dy = del[i * 4 + 1] * 0.5f;
    float dw = fminf(del[i * 4 + 2], SCALE_CLAMP);
    float dh = fminf(del[i * 4 + 3], SCALE_CLAMP);
    float pcx = dx * w + cx, pcy = dy * h + cy;
    float pw = __expf(dw) * w, ph = __expf(dh) * h;
    pred[i * 4 + 0] = pcx - 0.5f * pw;
    pred[i * 4 + 1] = pcy - 0.5f * ph;
    pred[i * 4 + 2] = pcx + 0.5f * pw;
    pred[i * 4 + 3] = pcy + 0.5f * ph;
}

// ---------------------------------------------------------------------------
// Host launcher
// ---------------------------------------------------------------------------
extern "C" void kernel_launch(void* const* d_in, const int* in_sizes, int n_in,
                              void* d_out, int out_size, void* d_ws, size_t ws_size,
                              hipStream_t stream) {
    (void)in_sizes; (void)n_in; (void)out_size; (void)ws_size;

    const float* pro       = (const float*)d_in[0];
    const float* roi       = (const float*)d_in[1];
    const float* bboxes    = (const float*)d_in[2];
    const float* attn_w_in = (const float*)d_in[3];
    const float* attn_b_in = (const float*)d_in[4];
    const float* attn_w_out= (const float*)d_in[5];
    const float* attn_b_out= (const float*)d_in[6];
    const float* ln1_g     = (const float*)d_in[7];
    const float* ln1_b     = (const float*)d_in[8];
    const float* dyn_w     = (const float*)d_in[9];
    const float* dyn_b     = (const float*)d_in[10];
    const float* dn1_g     = (const float*)d_in[11];
    const float* dn1_b     = (const float*)d_in[12];
    const float* dn2_g     = (const float*)d_in[13];
    const float* dn2_b     = (const float*)d_in[14];
    const float* out_w     = (const float*)d_in[15];
    const float* out_b     = (const float*)d_in[16];
    const float* dn3_g     = (const float*)d_in[17];
    const float* dn3_b     = (const float*)d_in[18];
    const float* ln2_g     = (const float*)d_in[19];
    const float* ln2_b     = (const float*)d_in[20];
    const float* ff1_w     = (const float*)d_in[21];
    const float* ff1_b     = (const float*)d_in[22];
    const float* ff2_w     = (const float*)d_in[23];
    const float* ff2_b     = (const float*)d_in[24];
    const float* ln3_g     = (const float*)d_in[25];
    const float* ln3_b     = (const float*)d_in[26];
    const float* cls_w     = (const float*)d_in[27];
    const float* cls_g     = (const float*)d_in[28];
    const float* cls_b     = (const float*)d_in[29];
    const float* reg_w     = (const float*)d_in[30];
    const float* reg_g     = (const float*)d_in[31];
    const float* reg_b     = (const float*)d_in[32];
    const float* logits_w  = (const float*)d_in[33];
    const float* logits_b  = (const float*)d_in[34];
    const float* delta_w   = (const float*)d_in[35];
    const float* delta_b   = (const float*)d_in[36];

    float* out        = (float*)d_out;
    float* logits_out = out;              // (16,100,2)  = 3200
    float* pred_out   = out + 3200;       // (16,100,4)  = 6400
    float* obj_out    = out + 9600;       // (1,1600,256)= 409600

    // workspace layout
    float* ws     = (float*)d_ws;
    float* x      = ws;                   // 409600  (pro permuted to (s,b,D))
    float* qkv    = x + 409600;           // 1228800
    float* o      = qkv + 1228800;        // 409600
    float* oproj  = o + 409600;           // 409600
    float* pro_ln = oproj + 409600;       // 409600
    float* fo     = pro_ln + 409600;      // 409600
    float* f3     = fo + 409600;          // 409600
    float* obj1   = f3 + 409600;          // 409600
    float* ffo    = obj1 + 409600;        // 409600
    float* towerA = ffo + 409600;         // 409600
    float* towerB = towerA + 409600;      // 409600
    float* towerC = towerB + 409600;      // 409600
    float* deltas = towerC + 409600;      // 6400
    unsigned short* dynC = (unsigned short*)(deltas + 6400); // 320*32768 bf16
    unsigned short* f2b  = dynC + 10485760;                  // 1600*49*256 bf16
    unsigned short* ffh  = f2b + 20070400;                   // 1600*2048 bf16

    // ---- self attention -------------------------------------------------
    permute_kernel<<<1600, 256, 0, stream>>>(pro, x);
    wmma_gemm<float, float, false><<<dim3(25, 12), 128, 0, stream>>>(
        x, attn_w_in, attn_b_in, qkv, 1600, 768, 256);
    attn_kernel<<<dim3(100, 8, 16), 128, 0, stream>>>(qkv, o);
    wmma_gemm<float, float, false><<<dim3(25, 4), 128, 0, stream>>>(
        o, attn_w_out, attn_b_out, oproj, 1600, 256, 256);
    ln_kernel<true, false, true><<<1600, 256, 0, stream>>>(oproj, x, ln1_g, ln1_b, pro_ln);

    // ---- dynamic conv, chunked over 5 x 320 instances -------------------
    for (int c = 0; c < 5; ++c) {
        wmma_gemm<float, unsigned short, false><<<dim3(5, 512), 128, 0, stream>>>(
            pro_ln + (size_t)c * 320 * 256, dyn_w, dyn_b, dynC, 320, 32768, 256);
        dynconv_kernel<<<320, 256, 0, stream>>>(
            roi, dynC, dn1_g, dn1_b, dn2_g, dn2_b, f2b, c * 320);
    }
    wmma_gemm<unsigned short, float, false><<<dim3(25, 4), 128, 0, stream>>>(
        f2b, out_w, out_b, fo, 1600, 256, 12544);
    ln_kernel<false, true, false><<<1600, 256, 0, stream>>>(fo, nullptr, dn3_g, dn3_b, f3);
    ln_kernel<true, false, false><<<1600, 256, 0, stream>>>(f3, pro_ln, ln2_g, ln2_b, obj1);

    // ---- FFN ------------------------------------------------------------
    wmma_gemm<float, unsigned short, true><<<dim3(25, 32), 128, 0, stream>>>(
        obj1, ff1_w, ff1_b, ffh, 1600, 2048, 256);
    wmma_gemm<unsigned short, float, false><<<dim3(25, 4), 128, 0, stream>>>(
        ffh, ff2_w, ff2_b, ffo, 1600, 256, 2048);
    ln_kernel<true, false, false><<<1600, 256, 0, stream>>>(ffo, obj1, ln3_g, ln3_b, obj_out);

    // ---- cls tower (1 layer) -------------------------------------------
    wmma_gemm<float, float, false><<<dim3(25, 4), 128, 0, stream>>>(
        obj_out, cls_w, nullptr, towerA, 1600, 256, 256);
    ln_kernel<false, true, false><<<1600, 256, 0, stream>>>(towerA, nullptr, cls_g, cls_b, towerB);

    // ---- reg tower (3 layers) ------------------------------------------
    const float* rin = obj_out;
    for (int i = 0; i < 3; ++i) {
        wmma_gemm<float, float, false><<<dim3(25, 4), 128, 0, stream>>>(
            rin, reg_w + (size_t)i * 65536, nullptr, towerA, 1600, 256, 256);
        ln_kernel<false, true, false><<<1600, 256, 0, stream>>>(
            towerA, nullptr, reg_g + i * 256, reg_b + i * 256, towerC);
        rin = towerC;
    }

    // ---- heads + box decode --------------------------------------------
    head_kernel<<<1600, 256, 0, stream>>>(towerB, logits_w, logits_b, logits_out, 2);
    head_kernel<<<1600, 256, 0, stream>>>(towerC, delta_w, delta_b, deltas, 4);
    apply_deltas_kernel<<<(1600 + 255) / 256, 256, 0, stream>>>(bboxes, deltas, pred_out);
}
